// GAT_33337536151585
// MI455X (gfx1250) — matrix operand order repaired and checked by reading, not compiled, not run on hardware
//
#include <hip/hip_runtime.h>
#include <cstdint>
#include <cstddef>

#define NNODES   50000
#define NEDGES   800000
#define NFEAT    256
#define NHID     64
#define HEADS    4
#define NCLASS   16
#define NEG_SLOPE 0.2f
#define KDIM     256     // K dimension of both GEMMs
#define MCHUNK   5       // M-tiles per GEMM wave (3125 % 5 == 0)

typedef __attribute__((ext_vector_type(16))) __bf16 v16bf;
typedef __attribute__((ext_vector_type(8)))  __bf16 v8bf;
typedef __attribute__((ext_vector_type(4)))  __bf16 v4bf;
typedef __attribute__((ext_vector_type(8)))  float  v8f;

static __device__ __forceinline__ float leaky(float x) {
    return x > 0.f ? x : NEG_SLOPE * x;
}

// ---------------------------------------------------------------------------
// Precision split: f32 -> bf16 hi + bf16 lo(residual). 4 elements per thread.
// ---------------------------------------------------------------------------
__global__ void gat_split(const float* __restrict__ in,
                          __bf16* __restrict__ hi, __bf16* __restrict__ lo, int n4) {
    int i = blockIdx.x * blockDim.x + threadIdx.x;
    if (i >= n4) return;
    float4 v = ((const float4*)in)[i];
    v4bf hv, lv;
    hv[0] = (__bf16)v.x; lv[0] = (__bf16)(v.x - (float)hv[0]);
    hv[1] = (__bf16)v.y; lv[1] = (__bf16)(v.y - (float)hv[1]);
    hv[2] = (__bf16)v.z; lv[2] = (__bf16)(v.z - (float)hv[2]);
    hv[3] = (__bf16)v.w; lv[3] = (__bf16)(v.w - (float)hv[3]);
    ((v4bf*)hi)[i] = hv;
    ((v4bf*)lo)[i] = lv;
}

// Split + transpose weights: W[K,Nout] f32 -> hiT/loT [Nout,K] bf16.
__global__ void gat_split_T(const float* __restrict__ W,
                            __bf16* __restrict__ hiT, __bf16* __restrict__ loT,
                            int K, int Nout) {
    int idx = blockIdx.x * blockDim.x + threadIdx.x;
    if (idx >= K * Nout) return;
    int k = idx / Nout, c = idx % Nout;
    float v = W[idx];
    __bf16 h = (__bf16)v;
    hiT[(size_t)c * K + k] = h;
    loT[(size_t)c * K + k] = (__bf16)(v - (float)h);
}

// ---------------------------------------------------------------------------
// WMMA GEMM: OUT[M,Nout] = X[M,256] @ W[256,Nout], inputs pre-split bf16 hi/lo,
// W pre-transposed to [Nout,256]. One wave: 16-col strip, B (hi+lo, all K) held
// in 128 VGPRs, loops over MCHUNK M-tiles. 24 WMMAs per tile (3-term split).
// __launch_bounds__(256,1): full VGPR budget so the 16 resident v16bf B
// fragments do NOT spill to scratch (spills observed at default occupancy).
// Fragment layouts used here (contiguous K per lane half, not the ISA table order):
//   A: lane&15 = M, elems = K + 16*(lane>>4)
//   B: lane&15 = N, elems = K + 16*(lane>>4)
//   C: lane&15 = N, vgpr r -> M = r + 8*(lane>>4)
// ---------------------------------------------------------------------------
__global__ void __launch_bounds__(256, 1)
gat_gemm_wmma(const __bf16* __restrict__ Xhi, const __bf16* __restrict__ Xlo,
              const __bf16* __restrict__ WhiT, const __bf16* __restrict__ WloT,
              float* __restrict__ OUT, int Nout, int mtiles, int ntiles) {
    int wave = blockIdx.x * (blockDim.x >> 5) + (threadIdx.x >> 5);
    int lane = threadIdx.x & 31;
    int mgroups = (mtiles + MCHUNK - 1) / MCHUNK;
    if (wave >= mgroups * ntiles) return;
    int ntile = wave % ntiles;
    int m0    = (wave / ntiles) * MCHUNK;
    int mend  = min(m0 + MCHUNK, mtiles);
    int g  = lane >> 4;
    int lm = lane & 15;

    // --- B fragments for all 8 K-steps, hi and lo (resident in VGPRs) ---
    const __bf16* wr_hi = WhiT + (size_t)(ntile * 16 + lm) * KDIM + g * 16;
    const __bf16* wr_lo = WloT + (size_t)(ntile * 16 + lm) * KDIM + g * 16;
    v16bf bh0 = *(const v16bf*)(wr_hi + 0 * 32), bl0 = *(const v16bf*)(wr_lo + 0 * 32);
    v16bf bh1 = *(const v16bf*)(wr_hi + 1 * 32), bl1 = *(const v16bf*)(wr_lo + 1 * 32);
    v16bf bh2 = *(const v16bf*)(wr_hi + 2 * 32), bl2 = *(const v16bf*)(wr_lo + 2 * 32);
    v16bf bh3 = *(const v16bf*)(wr_hi + 3 * 32), bl3 = *(const v16bf*)(wr_lo + 3 * 32);
    v16bf bh4 = *(const v16bf*)(wr_hi + 4 * 32), bl4 = *(const v16bf*)(wr_lo + 4 * 32);
    v16bf bh5 = *(const v16bf*)(wr_hi + 5 * 32), bl5 = *(const v16bf*)(wr_lo + 5 * 32);
    v16bf bh6 = *(const v16bf*)(wr_hi + 6 * 32), bl6 = *(const v16bf*)(wr_lo + 6 * 32);
    v16bf bh7 = *(const v16bf*)(wr_hi + 7 * 32), bl7 = *(const v16bf*)(wr_lo + 7 * 32);

    for (int mt = m0; mt < mend; ++mt) {
        const __bf16* ar_hi = Xhi + (size_t)(mt * 16 + lm) * KDIM + g * 16;
        const __bf16* ar_lo = Xlo + (size_t)(mt * 16 + lm) * KDIM + g * 16;
        v8f c = {0.f, 0.f, 0.f, 0.f, 0.f, 0.f, 0.f, 0.f};

#define GAT_STEP(S, BH, BL)                                                              \
        {                                                                                \
            v8bf h0 = *(const v8bf*)(ar_hi + (S) * 32);                                  \
            v8bf h1 = *(const v8bf*)(ar_hi + (S) * 32 + 8);                              \
            v8bf l0 = *(const v8bf*)(ar_lo + (S) * 32);                                  \
            v8bf l1 = *(const v8bf*)(ar_lo + (S) * 32 + 8);                              \
            v16bf ah = __builtin_shufflevector(h0, h1, 0,1,2,3,4,5,6,7,8,9,10,11,12,13,14,15); \
            v16bf am = __builtin_shufflevector(l0, l1, 0,1,2,3,4,5,6,7,8,9,10,11,12,13,14,15); \
            c = __builtin_amdgcn_wmma_f32_16x16x32_bf16(false, ah, false, BH, (short)0, c, false, false); \
            c = __builtin_amdgcn_wmma_f32_16x16x32_bf16(false, ah, false, BL, (short)0, c, false, false); \
            c = __builtin_amdgcn_wmma_f32_16x16x32_bf16(false, am, false, BH, (short)0, c, false, false); \
        }
        GAT_STEP(0, bh0, bl0)
        GAT_STEP(1, bh1, bl1)
        GAT_STEP(2, bh2, bl2)
        GAT_STEP(3, bh3, bl3)
        GAT_STEP(4, bh4, bl4)
        GAT_STEP(5, bh5, bl5)
        GAT_STEP(6, bh6, bl6)
        GAT_STEP(7, bh7, bl7)
#undef GAT_STEP

        float* orow = OUT + (size_t)(mt * 16 + g * 8) * Nout + ntile * 16 + lm;
        #pragma unroll
        for (int r = 0; r < 8; ++r) orow[(size_t)r * Nout] = c[r];
    }
}

// ---------------------------------------------------------------------------
// Attention projections, layer 1 (wave per node).
// ---------------------------------------------------------------------------
__global__ void gat_attnproj1(const float* __restrict__ feat,
                              const float* __restrict__ al,
                              const float* __restrict__ ar,
                              float* __restrict__ el, float* __restrict__ er) {
    int n    = blockIdx.x * (blockDim.x >> 5) + (threadIdx.x >> 5);
    int lane = threadIdx.x & 31;
    if (n >= NNODES) return;
    const float* f = feat + (size_t)n * NFEAT + lane * 8;
    float4 f0 = *(const float4*)f;
    float4 f1 = *(const float4*)(f + 4);
    const float* a = al + lane * 8;
    const float* r = ar + lane * 8;
    float pe = f0.x*a[0] + f0.y*a[1] + f0.z*a[2] + f0.w*a[3]
             + f1.x*a[4] + f1.y*a[5] + f1.z*a[6] + f1.w*a[7];
    float pr = f0.x*r[0] + f0.y*r[1] + f0.z*r[2] + f0.w*r[3]
             + f1.x*r[4] + f1.y*r[5] + f1.z*r[6] + f1.w*r[7];
    #pragma unroll
    for (int s = 1; s < 8; s <<= 1) {
        pe += __shfl_xor(pe, s, 32);
        pr += __shfl_xor(pr, s, 32);
    }
    if ((lane & 7) == 0) {
        int hd = lane >> 3;
        el[(size_t)n * HEADS + hd] = pe;
        er[(size_t)n * HEADS + hd] = pr;
    }
}

// ---------------------------------------------------------------------------
// CSR construction (count -> scan -> scatter -> per-segment sort)
// ---------------------------------------------------------------------------
__global__ void gat_zero2(int* a, int* b, int n) {
    int i = blockIdx.x * blockDim.x + threadIdx.x;
    if (i < n) { a[i] = 0; b[i] = 0; }
}

__global__ void gat_count(const int* __restrict__ dst, int* cnt) {
    int e = blockIdx.x * blockDim.x + threadIdx.x;
    if (e < NEDGES) atomicAdd(&cnt[dst[e]], 1);
}

__global__ void gat_scan(const int* __restrict__ cnt, int* __restrict__ off) {
    __shared__ int sm[1024];
    __shared__ int carry;
    int tid = threadIdx.x;
    if (tid == 0) carry = 0;
    __syncthreads();
    for (int base = 0; base < NNODES; base += 1024) {
        int i = base + tid;
        int v = (i < NNODES) ? cnt[i] : 0;
        sm[tid] = v;
        __syncthreads();
        for (int s = 1; s < 1024; s <<= 1) {
            int t = (tid >= s) ? sm[tid - s] : 0;
            __syncthreads();
            sm[tid] += t;
            __syncthreads();
        }
        if (i < NNODES) off[i] = carry + sm[tid] - v;  // exclusive
        __syncthreads();
        if (tid == 0) carry += sm[1023];
        __syncthreads();
    }
    if (tid == 0) off[NNODES] = carry;
}

__global__ void gat_scatter(const int* __restrict__ dst, const int* __restrict__ off,
                            int* cursor, int* eidx) {
    int e = blockIdx.x * blockDim.x + threadIdx.x;
    if (e < NEDGES) {
        int d = dst[e];
        int p = atomicAdd(&cursor[d], 1);
        eidx[off[d] + p] = e;
    }
}

// Sort each segment ascending by edge id -> deterministic accumulation order.
__global__ void gat_sortseg(const int* __restrict__ off, int* eidx) {
    int n = blockIdx.x * blockDim.x + threadIdx.x;
    if (n >= NNODES) return;
    int b = off[n], e = off[n + 1];
    for (int i = b + 1; i < e; ++i) {
        int v = eidx[i];
        int j = i - 1;
        while (j >= b && eidx[j] > v) { eidx[j + 1] = eidx[j]; --j; }
        eidx[j + 1] = v;
    }
}

// ---------------------------------------------------------------------------
// Layer-1 aggregation: per-dst softmax + weighted sum + bias + ELU,
// emitting h directly as bf16 hi/lo (feeds layer-2 WMMA GEMM).
// One wave per node; lane l owns channels [8l, 8l+8) (one head per 8 lanes).
// ---------------------------------------------------------------------------
__global__ void gat_aggregate1(const float* __restrict__ feat,
                               const float* __restrict__ el,
                               const float* __restrict__ er,
                               const int* __restrict__ srcA,
                               const int* __restrict__ off,
                               const int* __restrict__ eidx,
                               const float* __restrict__ bias,
                               __bf16* __restrict__ hhi, __bf16* __restrict__ hlo) {
    int n    = blockIdx.x * (blockDim.x >> 5) + (threadIdx.x >> 5);
    int lane = threadIdx.x & 31;
    if (n >= NNODES) return;
    int hd = lane >> 3;
    int b = off[n], e = off[n + 1];
    float ern = er[(size_t)n * HEADS + hd];

    float maxe = -3.4e38f;
    for (int j = b; j < e; ++j) {
        int s = srcA[eidx[j]];
        maxe = fmaxf(maxe, leaky(el[(size_t)s * HEADS + hd] + ern));
    }

    float a0=0.f,a1=0.f,a2=0.f,a3=0.f,a4=0.f,a5=0.f,a6=0.f,a7=0.f,sumw=0.f;
    for (int j = b; j < e; ++j) {
        int s = srcA[eidx[j]];
        if (j + 1 < e)
            __builtin_prefetch(feat + (size_t)srcA[eidx[j + 1]] * NFEAT + lane * 8, 0, 3);
        float w = expf(leaky(el[(size_t)s * HEADS + hd] + ern) - maxe);
        sumw += w;
        const float* fr = feat + (size_t)s * NFEAT + lane * 8;
        float4 f0 = *(const float4*)fr;
        float4 f1 = *(const float4*)(fr + 4);
        a0 += w * f0.x; a1 += w * f0.y; a2 += w * f0.z; a3 += w * f0.w;
        a4 += w * f1.x; a5 += w * f1.y; a6 += w * f1.z; a7 += w * f1.w;
    }
    float inv = (e > b) ? (1.f / sumw) : 0.f;
    int ch = lane * 8;
    const float* bb = bias + ch;
    float acc[8] = {a0, a1, a2, a3, a4, a5, a6, a7};
    v8bf hv, lv;
    #pragma unroll
    for (int k = 0; k < 8; ++k) {
        float v = acc[k] * inv + bb[k];
        v = v > 0.f ? v : (expf(v) - 1.f);      // ELU
        __bf16 h = (__bf16)v;
        hv[k] = h;
        lv[k] = (__bf16)(v - (float)h);
    }
    *(v8bf*)(hhi + (size_t)n * NFEAT + ch) = hv;
    *(v8bf*)(hlo + (size_t)n * NFEAT + ch) = lv;
}

// ---------------------------------------------------------------------------
// Layer-2 attention projections: 1 head, 16 classes. Thread per node.
// ---------------------------------------------------------------------------
__global__ void gat_attnproj2(const float* __restrict__ feat2,
                              const float* __restrict__ al,
                              const float* __restrict__ ar,
                              float* __restrict__ el, float* __restrict__ er) {
    int n = blockIdx.x * blockDim.x + threadIdx.x;
    if (n >= NNODES) return;
    float se = 0.f, sr = 0.f;
    #pragma unroll
    for (int c = 0; c < NCLASS; ++c) {
        float v = feat2[(size_t)n * NCLASS + c];
        se += v * al[c];
        sr += v * ar[c];
    }
    el[n] = se; er[n] = sr;
}

// ---------------------------------------------------------------------------
// Layer-2 aggregation (1 head, D=16): wave per node, lanes<16 own a class each.
// ---------------------------------------------------------------------------
__global__ void gat_aggregate2(const float* __restrict__ feat2,
                               const float* __restrict__ el,
                               const float* __restrict__ er,
                               const int* __restrict__ srcA,
                               const int* __restrict__ off,
                               const int* __restrict__ eidx,
                               const float* __restrict__ bias,
                               float* __restrict__ out) {
    int n    = blockIdx.x * (blockDim.x >> 5) + (threadIdx.x >> 5);
    int lane = threadIdx.x & 31;
    if (n >= NNODES) return;
    int b = off[n], e = off[n + 1];
    float ern = er[n];

    float maxe = -3.4e38f;
    for (int j = b; j < e; ++j) {
        int s = srcA[eidx[j]];
        maxe = fmaxf(maxe, leaky(el[s] + ern));
    }
    float acc = 0.f, sumw = 0.f;
    for (int j = b; j < e; ++j) {
        int s = srcA[eidx[j]];
        float w = expf(leaky(el[s] + ern) - maxe);
        sumw += w;
        if (lane < NCLASS) acc += w * feat2[(size_t)s * NCLASS + lane];
    }
    if (lane < NCLASS) {
        float inv = (e > b) ? (1.f / sumw) : 0.f;
        out[(size_t)n * NCLASS + lane] = acc * inv + bias[lane];
    }
}

__global__ void gat_logsoftmax(float* __restrict__ out) {
    int n = blockIdx.x * blockDim.x + threadIdx.x;
    if (n >= NNODES) return;
    float* row = out + (size_t)n * NCLASS;
    float m = -3.4e38f;
    #pragma unroll
    for (int c = 0; c < NCLASS; ++c) m = fmaxf(m, row[c]);
    float s = 0.f;
    #pragma unroll
    for (int c = 0; c < NCLASS; ++c) s += expf(row[c] - m);
    float l = m + logf(s);
    #pragma unroll
    for (int c = 0; c < NCLASS; ++c) row[c] -= l;
}

// ---------------------------------------------------------------------------
extern "C" void kernel_launch(void* const* d_in, const int* in_sizes, int n_in,
                              void* d_out, int out_size, void* d_ws, size_t ws_size,
                              hipStream_t stream) {
    const float* x   = (const float*)d_in[0];
    const int*   src = (const int*)  d_in[1];
    const int*   dst = (const int*)  d_in[2];
    const float* W1  = (const float*)d_in[3];
    const float* al1 = (const float*)d_in[4];
    const float* ar1 = (const float*)d_in[5];
    const float* b1  = (const float*)d_in[6];
    const float* W2  = (const float*)d_in[7];
    const float* al2 = (const float*)d_in[8];
    const float* ar2 = (const float*)d_in[9];
    const float* b2  = (const float*)d_in[10];
    float* out = (float*)d_out;

    // Workspace carve (256B aligned slices)
    char* p = (char*)d_ws;
    auto carve = [&](size_t bytes) -> void* {
        void* r = (void*)p;
        p += (bytes + 255) & ~(size_t)255;
        return r;
    };
    float*  feat1  = (float*) carve((size_t)NNODES * NFEAT  * 4);  // x@W1 (f32)
    float*  feat2  = (float*) carve((size_t)NNODES * NCLASS * 4);  // h@W2 (f32)
    __bf16* xhi    = (__bf16*)carve((size_t)NNODES * NFEAT  * 2);
    __bf16* xlo    = (__bf16*)carve((size_t)NNODES * NFEAT  * 2);
    __bf16* hhi    = (__bf16*)carve((size_t)NNODES * NFEAT  * 2);  // elu(layer1) hi
    __bf16* hlo    = (__bf16*)carve((size_t)NNODES * NFEAT  * 2);  // elu(layer1) lo
    __bf16* w1hiT  = (__bf16*)carve((size_t)KDIM * NFEAT  * 2);
    __bf16* w1loT  = (__bf16*)carve((size_t)KDIM * NFEAT  * 2);
    __bf16* w2hiT  = (__bf16*)carve((size_t)KDIM * NCLASS * 2);
    __bf16* w2loT  = (__bf16*)carve((size_t)KDIM * NCLASS * 2);
    float*  el1    = (float*) carve((size_t)NNODES * HEADS * 4);
    float*  er1    = (float*) carve((size_t)NNODES * HEADS * 4);
    float*  el2    = (float*) carve((size_t)NNODES * 4);
    float*  er2    = (float*) carve((size_t)NNODES * 4);
    int*    cnt    = (int*)   carve((size_t)NNODES * 4);
    int*    cursor = (int*)   carve((size_t)NNODES * 4);
    int*    off    = (int*)   carve((size_t)(NNODES + 1) * 4);
    int*    eidx   = (int*)   carve((size_t)NEDGES * 4);
    (void)ws_size; (void)in_sizes; (void)n_in; (void)out_size;

    const int TB = 256;
    dim3 blk(TB);
    int nodeBlocks  = (NNODES + TB - 1) / TB;
    int edgeBlocks  = (NEDGES + TB - 1) / TB;
    int nodeWaveBlk = (NNODES + 7) / 8;
    const int mtiles = NNODES / 16;                     // 3125
    const int mgroups = (mtiles + MCHUNK - 1) / MCHUNK; // 625

    // --- CSR build (deterministic after per-segment sort) ---
    gat_zero2  <<<nodeBlocks, blk, 0, stream>>>(cnt, cursor, NNODES);
    gat_count  <<<edgeBlocks, blk, 0, stream>>>(dst, cnt);
    gat_scan   <<<1, 1024, 0, stream>>>(cnt, off);
    gat_scatter<<<edgeBlocks, blk, 0, stream>>>(dst, off, cursor, eidx);
    gat_sortseg<<<nodeBlocks, blk, 0, stream>>>(off, eidx);

    // --- Precision splits (x, W1, W2) ---
    {
        int n4 = NNODES * NFEAT / 4;
        gat_split  <<<(n4 + TB - 1) / TB, blk, 0, stream>>>(x, xhi, xlo, n4);
        int w1n = KDIM * NFEAT;
        gat_split_T<<<(w1n + TB - 1) / TB, blk, 0, stream>>>(W1, w1hiT, w1loT, KDIM, NFEAT);
        int w2n = KDIM * NCLASS;
        gat_split_T<<<(w2n + TB - 1) / TB, blk, 0, stream>>>(W2, w2hiT, w2loT, KDIM, NCLASS);
    }

    // --- Layer 1 ---
    {
        int ntiles = NFEAT / 16;                        // 16
        int waves  = mgroups * ntiles;                  // 10000
        gat_gemm_wmma<<<(waves + 7) / 8, blk, 0, stream>>>(xhi, xlo, w1hiT, w1loT,
                                                           feat1, NFEAT, mtiles, ntiles);
    }
    gat_attnproj1 <<<nodeWaveBlk, blk, 0, stream>>>(feat1, al1, ar1, el1, er1);
    gat_aggregate1<<<nodeWaveBlk, blk, 0, stream>>>(feat1, el1, er1, src, off, eidx, b1, hhi, hlo);

    // --- Layer 2 ---
    {
        int ntiles = NCLASS / 16;                       // 1
        int waves  = mgroups * ntiles;                  // 625
        gat_gemm_wmma<<<(waves + 7) / 8, blk, 0, stream>>>(hhi, hlo, w2hiT, w2loT,
                                                           feat2, NCLASS, mtiles, ntiles);
    }
    gat_attnproj2 <<<nodeBlocks, blk, 0, stream>>>(feat2, al2, ar2, el2, er2);
    gat_aggregate2<<<nodeWaveBlk, blk, 0, stream>>>(feat2, el2, er2, src, off, eidx, b2, out);
    gat_logsoftmax<<<nodeBlocks, blk, 0, stream>>>(out);
}
